// DeepseekV3TopkRouter_62989990363213
// MI455X (gfx1250) — compile-verified, matrix operand order checked by reading, not run on hardware
//
#include <hip/hip_runtime.h>
#include <hip/hip_bf16.h>

// DeepSeek-V3 router on gfx1250 (MI455X):
//   Kernel 0: W fp32 -> bf16 once (3.67 MB, L2-resident for the whole GEMM)
//   Kernel 1: logits = h @ W^T via V_WMMA_F32_16X16X32_BF16, fragments loaded
//             straight from global memory in ISA 7.12.2 layout (no LDS, no
//             barriers); A packed fp32->bf16 with one V_PERM_B32 per pair.
//   Kernel 2: per-token grouped top-k routing.

#define T_TOKENS   16384
#define HIDDEN_DIM 7168
#define N_EXP      256
#define GROUP_SZ   32
#define N_GRP      8
#define TOPK_GRP   4
#define TOP_K      8
#define ROUTED_SCALING 2.5f

typedef __attribute__((ext_vector_type(16))) __bf16 v16bf;
typedef __attribute__((ext_vector_type(8)))  __bf16 v8bf;
typedef __attribute__((ext_vector_type(8)))  float  v8f;

// Pack two fp32 into bf16x2 (truncation) with a single byte-permute:
// dst[15:0] = lo[31:16], dst[31:16] = hi[31:16].
__device__ __forceinline__ unsigned pkbf(float lo, float hi) {
    union { float f; unsigned u; } L, H; L.f = lo; H.f = hi;
#if __has_builtin(__builtin_amdgcn_perm)
    return __builtin_amdgcn_perm(H.u, L.u, 0x07060302u);  // v_perm_b32
#else
    return (H.u & 0xffff0000u) | (L.u >> 16);
#endif
}

union AFrag { v16bf v; unsigned u[8]; };
union BFrag { v16bf v; v8bf h[2]; };

// ---------------------------------------------------------------------------
// Kernel 0: convert W (fp32 [256][7168]) to bf16 in workspace.
// ---------------------------------------------------------------------------
__global__ __launch_bounds__(256)
void convert_w_kernel(const float* __restrict__ w, unsigned* __restrict__ wb_u32) {
    const size_t i4 = (size_t)blockIdx.x * 256 + threadIdx.x; // float4 index
    const float4 v = ((const float4*)w)[i4];
    uint2 pk;
    pk.x = pkbf(v.x, v.y);
    pk.y = pkbf(v.z, v.w);
    ((uint2*)wb_u32)[i4] = pk;
}

// ---------------------------------------------------------------------------
// Kernel 1: router logits GEMM. Grid = T/64 blocks, 256 threads (8 waves).
// Block: 64 tokens x 256 experts. Wave w: M-tile = w/2, N-tiles (w%2)*8..+7.
// Fragment layout (ISA 7.12.2, 16-bit A/B 16x32): lane l = {half = l/16,
// idx = l%16}; elems 0..7 -> K = half*8..+7, elems 8..15 -> K = 16+half*8..+7.
// Both chunks are 16B-contiguous in row-major memory -> direct global_load_b128.
// ---------------------------------------------------------------------------
__global__ __launch_bounds__(256)
void router_gemm_kernel(const float* __restrict__ h,
                        const __bf16* __restrict__ wb,
                        float* __restrict__ logits) {
    const int tid  = threadIdx.x;
    const int lane = tid & 31;
    const int wave = tid >> 5;
    const int m0   = blockIdx.x * 64;

    const int mt   = wave >> 1;        // 0..3 (M-tile of 16 tokens)
    const int ntb  = (wave & 1) * 8;   // base N-tile (16 experts per tile)
    const int half = lane >> 4;
    const int ml   = lane & 15;

    // per-lane streaming pointers
    const float*  aptr = h  + (size_t)(m0 + mt * 16 + ml) * HIDDEN_DIM + half * 8;
    const __bf16* bptr = wb + (size_t)(ntb * 16 + ml)     * HIDDEN_DIM + half * 8;

    v8f acc[8];
    #pragma unroll
    for (int j = 0; j < 8; ++j)
        #pragma unroll
        for (int r = 0; r < 8; ++r) acc[j][r] = 0.0f;

    for (int kk = 0; kk < HIDDEN_DIM; kk += 32) {
        // ---- A fragment: 16 fp32 -> 8 bf16x2 dwords via v_perm_b32 ----
        const float4 a0 = *(const float4*)(aptr + 0);
        const float4 a1 = *(const float4*)(aptr + 4);
        const float4 a2 = *(const float4*)(aptr + 16);
        const float4 a3 = *(const float4*)(aptr + 20);
        AFrag af;
        af.u[0] = pkbf(a0.x, a0.y); af.u[1] = pkbf(a0.z, a0.w);
        af.u[2] = pkbf(a1.x, a1.y); af.u[3] = pkbf(a1.z, a1.w);
        af.u[4] = pkbf(a2.x, a2.y); af.u[5] = pkbf(a2.z, a2.w);
        af.u[6] = pkbf(a3.x, a3.y); af.u[7] = pkbf(a3.z, a3.w);

        // ---- 8 B fragments straight from bf16 W (L2-resident) + 8 WMMAs ----
        // N-tile j sits at byte offset j*16*7168*2 = j*229376 (fits 24-bit IOFFSET)
        #pragma unroll
        for (int j = 0; j < 8; ++j) {
            const __bf16* bj = bptr + (size_t)j * 16 * HIDDEN_DIM;
            BFrag bf;
            bf.h[0] = *(const v8bf*)(bj + 0);
            bf.h[1] = *(const v8bf*)(bj + 16);
            acc[j] = __builtin_amdgcn_wmma_f32_16x16x32_bf16(
                /*neg_a=*/false, af.v, /*neg_b=*/false, bf.v,
                /*c_mod=*/(short)0, acc[j],
                /*reuse_a=*/false, /*reuse_b=*/false);
        }
        aptr += 32;
        bptr += 32;
    }

    // ---- epilogue: C/D layout: VGPR r -> M = r + 8*half, N = lane%16 ----
    const int mbase = m0 + mt * 16 + half * 8;
    #pragma unroll
    for (int j = 0; j < 8; ++j) {
        const int n = (ntb + j) * 16 + ml;
        #pragma unroll
        for (int r = 0; r < 8; ++r)
            logits[(size_t)(mbase + r) * N_EXP + n] = acc[j][r];
    }
}

// ---------------------------------------------------------------------------
// Kernel 2: per-token grouped top-k routing. One thread per token.
// ---------------------------------------------------------------------------
__global__ __launch_bounds__(256)
void router_topk_kernel(const float* __restrict__ logits,
                        const float* __restrict__ bias,
                        int* __restrict__ out_idx,
                        float* __restrict__ out_w) {
    const int t = blockIdx.x * blockDim.x + threadIdx.x;
    if (t >= T_TOKENS) return;
    const float4* lrow4 = (const float4*)(logits + (size_t)t * N_EXP);
    const float4* bias4 = (const float4*)bias;

    // pass 1: group score = sum of top-2 corrected scores in each group of 32
    float gscore[N_GRP];
    #pragma unroll
    for (int g = 0; g < N_GRP; ++g) {
        float t1 = -1e30f, t2 = -1e30f;
        for (int i = 0; i < GROUP_SZ / 4; ++i) {
            const float4 lv = lrow4[g * 8 + i];
            const float4 bv = bias4[g * 8 + i];
            float s[4];
            s[0] = 1.0f / (1.0f + __expf(-lv.x)) + bv.x;
            s[1] = 1.0f / (1.0f + __expf(-lv.y)) + bv.y;
            s[2] = 1.0f / (1.0f + __expf(-lv.z)) + bv.z;
            s[3] = 1.0f / (1.0f + __expf(-lv.w)) + bv.w;
            #pragma unroll
            for (int c = 0; c < 4; ++c) {
                if (s[c] > t1)      { t2 = t1; t1 = s[c]; }
                else if (s[c] > t2) { t2 = s[c]; }
            }
        }
        gscore[g] = t1 + t2;
    }

    // top-4 groups (selection)
    int sel[TOPK_GRP];
    #pragma unroll
    for (int j = 0; j < TOPK_GRP; ++j) {
        int best = 0; float bv = -1e30f;
        #pragma unroll
        for (int g = 0; g < N_GRP; ++g)
            if (gscore[g] > bv) { bv = gscore[g]; best = g; }
        sel[j] = best;
        gscore[best] = -1e30f;
    }

    // pass 2: top-8 experts among the 4 selected groups (register insertion)
    float tv[TOP_K], tw[TOP_K]; int ti[TOP_K];
    #pragma unroll
    for (int j = 0; j < TOP_K; ++j) { tv[j] = -1e30f; tw[j] = 0.0f; ti[j] = 0; }

    for (int j = 0; j < TOPK_GRP; ++j) {
        const int g = sel[j];
        for (int i = 0; i < GROUP_SZ / 4; ++i) {
            const float4 lv = lrow4[g * 8 + i];
            const float4 bv = bias4[g * 8 + i];
            float raw[4];
            raw[0] = 1.0f / (1.0f + __expf(-lv.x));
            raw[1] = 1.0f / (1.0f + __expf(-lv.y));
            raw[2] = 1.0f / (1.0f + __expf(-lv.z));
            raw[3] = 1.0f / (1.0f + __expf(-lv.w));
            const float bb[4] = { bv.x, bv.y, bv.z, bv.w };
            #pragma unroll
            for (int c = 0; c < 4; ++c) {
                float cv = raw[c] + bb[c];     // corrected (selection key)
                float cw = raw[c];             // uncorrected (routing weight)
                int   ci = g * GROUP_SZ + i * 4 + c;
                #pragma unroll
                for (int k = 0; k < TOP_K; ++k) {
                    if (cv > tv[k]) {
                        float a = tv[k]; tv[k] = cv; cv = a;
                        float b = tw[k]; tw[k] = cw; cw = b;
                        int   d = ti[k]; ti[k] = ci; ci = d;
                    }
                }
            }
        }
    }

    float denom = 1e-20f;
    #pragma unroll
    for (int j = 0; j < TOP_K; ++j) denom += tw[j];
    const float scale = ROUTED_SCALING / denom;

    #pragma unroll
    for (int j = 0; j < TOP_K; ++j) {
        out_idx[t * TOP_K + j] = ti[j];
        out_w[t * TOP_K + j]   = tw[j] * scale;
    }
}

// ---------------------------------------------------------------------------
extern "C" void kernel_launch(void* const* d_in, const int* in_sizes, int n_in,
                              void* d_out, int out_size, void* d_ws, size_t ws_size,
                              hipStream_t stream) {
    const float* h    = (const float*)d_in[0];  // [4,4096,7168] f32
    const float* w    = (const float*)d_in[1];  // [256,7168]    f32
    const float* bias = (const float*)d_in[2];  // [256]         f32

    float* out      = (float*)d_out;
    int*   out_idx  = (int*)out;                          // [T,8] int32 (bit-cast slot)
    float* out_w    = out + (size_t)T_TOKENS * TOP_K;     // [T,8] f32
    float* logits   = out + (size_t)T_TOKENS * 2 * TOP_K; // [T,256] f32

    __bf16* wb = (__bf16*)d_ws;                           // 256*7168*2 = 3.67 MB scratch

    // W fp32 -> bf16 (458752 float4 chunks / 256 threads = 1792 blocks)
    convert_w_kernel<<<dim3((N_EXP * HIDDEN_DIM / 4) / 256), dim3(256), 0, stream>>>(w, (unsigned*)wb);
    router_gemm_kernel<<<dim3(T_TOKENS / 64), dim3(256), 0, stream>>>(h, wb, logits);
    router_topk_kernel<<<dim3(T_TOKENS / 256), dim3(256), 0, stream>>>(logits, bias, out_idx, out_w);
}
